// ArtAttention_57028575756695
// MI455X (gfx1250) — matrix operand order repaired and checked by reading, not compiled, not run on hardware
//
#include <hip/hip_runtime.h>
#include <hip/hip_bf16.h>

typedef __attribute__((ext_vector_type(16))) _Float16 v16h;
typedef __attribute__((ext_vector_type(8)))  _Float16 h8;
typedef __attribute__((ext_vector_type(8)))  float    v8f;

#define BB 4
#define TT 512
#define MMTOK 77
#define HH 10
#define DDIM 64
#define LAT 640
#define EE 8
#define FFN 256
#define TED 512
#define NKPAD 704          // 512 motion + 48 dataset + 48 rotation + 77 text + 19 pad
#define NKTILES 22         // 704 / 32
#define NROWS_M (BB*TT*HH)        // 20480
#define NROWS_T (BB*MMTOK*HH)     // 3080
#define NROWS_T_PAD 3088

// ---------------------------------------------------------------- helpers ----

__device__ __forceinline__ v8f wmma16(v16h a, v16h b, v8f c) {
  return __builtin_amdgcn_wmma_f32_16x16x32_f16(false, a, false, b, (short)0, c, false, false);
}

__device__ __forceinline__ v16h cat8(h8 lo, h8 hi) {
  return __builtin_shufflevector(lo, hi, 0,1,2,3,4,5,6,7,8,9,10,11,12,13,14,15);
}

// A operand (16x32 f16), source row-major, row = lane&15 (+ caller base), K window at koff.
// k order used here (contiguous, matches B; not the ISA table): lanes0-15 K=0..15, lanes16-31 K=16..31
__device__ __forceinline__ v16h load_a16(const _Float16* base, int stride, int koff) {
  int lane = threadIdx.x & 31; int lh = lane >> 4; int m = lane & 15;
  const _Float16* p = base + (long)m * stride + koff + lh * 16;
  h8 lo = *(const h8*)(p);
  h8 hi = *(const h8*)(p + 8);
  return cat8(lo, hi);
}

// B operand (32x16 f16) from "N-major" storage: row index = B column n = rowoff + lane&15,
// contiguous dimension = K.  k order used here (contiguous, matches A; not the ISA table): lanes0-15 K=0..15, lanes16-31 K=16..31
__device__ __forceinline__ v16h load_b16(const _Float16* base, int stride, int rowoff, int koff) {
  int lane = threadIdx.x & 31; int lh = lane >> 4; int n = lane & 15;
  const _Float16* p = base + (long)(rowoff + n) * stride + koff + lh * 16;
  h8 lo = *(const h8*)(p);
  h8 hi = *(const h8*)(p + 8);
  return cat8(lo, hi);
}

// B operand from pre-swizzled weight blob: 512 f16 per 32x16 tile, 16 contiguous per lane.
__device__ __forceinline__ v16h load_bpacked(const _Float16* blob) {
  const _Float16* p = blob + (threadIdx.x & 31) * 16;
  h8 lo = *(const h8*)(p);
  h8 hi = *(const h8*)(p + 8);
  return cat8(lo, hi);
}

// Async global -> LDS copy of one 16-byte chunk per lane (ASYNCcnt path).
__device__ __forceinline__ void async_copy_b128(const _Float16* gsrc, _Float16* lds_dst) {
  unsigned int loff = (unsigned int)(uintptr_t)lds_dst;   // low 32 bits = LDS byte address
  unsigned long long gaddr = (unsigned long long)(uintptr_t)gsrc;
  asm volatile("global_load_async_to_lds_b128 %0, %1, off"
               :: "v"(loff), "v"(gaddr) : "memory");
}
__device__ __forceinline__ void async_wait0() {
  asm volatile("s_wait_asynccnt 0x0" ::: "memory");
}

__device__ __forceinline__ float gelu_f(float x) {
  float x3 = x * x * x;
  return 0.5f * x * (1.0f + tanhf(0.7978845608028654f * (x + 0.044715f * x3)));
}
__device__ __forceinline__ float silu_f(float x) { return x / (1.0f + expf(-x)); }

__device__ __forceinline__ float redmax16(float v) {
  for (int m = 1; m < 16; m <<= 1) v = fmaxf(v, __shfl_xor(v, m, 32));
  return v;
}
__device__ __forceinline__ float redsum16(float v) {
  for (int m = 1; m < 16; m <<= 1) v += __shfl_xor(v, m, 32);
  return v;
}

__device__ __forceinline__ float block_reduce(float v, float* red) {
  int t = threadIdx.x;
  red[t] = v; __syncthreads();
  for (int off = 128; off > 0; off >>= 1) {
    if (t < off) red[t] += red[t + off];
    __syncthreads();
  }
  float r = red[0]; __syncthreads();
  return r;
}

// ------------------------------------------------------------- pack weights ---
// f32 row-major (K x N) -> f16 B-operand blobs: blob(s,nt) covers K rows [32s,32s+32),
// N cols [16nt,16nt+16).  Within blob: lane l, element j -> k = 32s + (l>>4)*16 + j,
// n = nt*16 + (l&15).
__global__ void pack_b_kernel(const float* __restrict__ src, _Float16* __restrict__ dst,
                              int K, int N, int batch, long sStride) {
  int KT = K >> 5, NT = N >> 4;
  long total = (long)batch * KT * NT * 512;
  long stride = (long)gridDim.x * blockDim.x;
  for (long i = (long)blockIdx.x * blockDim.x + threadIdx.x; i < total; i += stride) {
    int j = (int)(i & 15);
    int l = (int)((i >> 4) & 31);
    long blob = i >> 9;
    long e = blob / (KT * NT);
    int bl = (int)(blob % (KT * NT));
    int s = bl / NT, nt = bl % NT;
    int k = 32 * s + (l >> 4) * 16 + j;
    int n = nt * 16 + (l & 15);
    dst[i] = (_Float16)src[e * sStride + (long)k * N + n];
  }
}

// ------------------------------------------------------------------ LN -------
__global__ void ln_kernel(const float* __restrict__ x, const float* __restrict__ g,
                          const float* __restrict__ bta, const float* __restrict__ addemb,
                          float* __restrict__ out32, _Float16* __restrict__ out16,
                          int C, int Tmod) {
  __shared__ float red[256];
  long tok = blockIdx.x;
  const float* xr = x + tok * C;
  float s = 0.f;
  for (int j = threadIdx.x; j < C; j += 256) s += xr[j];
  float mean = block_reduce(s, red) / (float)C;
  float s2 = 0.f;
  for (int j = threadIdx.x; j < C; j += 256) { float d = xr[j] - mean; s2 += d * d; }
  float var = block_reduce(s2, red) / (float)C;
  float rs = rsqrtf(var + 1e-5f);
  long ebase = (tok % Tmod) * C;
  for (int j = threadIdx.x; j < C; j += 256) {
    float v = (xr[j] - mean) * rs * g[j] + bta[j];
    if (addemb) v += addemb[ebase + j];
    out32[tok * C + j] = v;
    out16[tok * C + j] = (_Float16)v;
  }
}

// modulated LN for the output path: silu(LN(x)*(1+eo[:640]) + eo[640:]) -> f16
__global__ void ln_mod_kernel(const float* __restrict__ x, const float* __restrict__ g,
                              const float* __restrict__ bta, const float* __restrict__ eo,
                              _Float16* __restrict__ out16) {
  __shared__ float red[256];
  long tok = blockIdx.x;
  const float* xr = x + tok * LAT;
  float s = 0.f;
  for (int j = threadIdx.x; j < LAT; j += 256) s += xr[j];
  float mean = block_reduce(s, red) / (float)LAT;
  float s2 = 0.f;
  for (int j = threadIdx.x; j < LAT; j += 256) { float d = xr[j] - mean; s2 += d * d; }
  float var = block_reduce(s2, red) / (float)LAT;
  float rs = rsqrtf(var + 1e-5f);
  for (int j = threadIdx.x; j < LAT; j += 256) {
    float v = (xr[j] - mean) * rs * g[j] + bta[j];
    v = v * (1.0f + eo[tok * 1280 + j]) + eo[tok * 1280 + 640 + j];
    out16[tok * LAT + j] = (_Float16)silu_f(v);
  }
}

__global__ void zero_pad_kernel(_Float16* xt16, float* combt) {
  int i = threadIdx.x;                 // 512 threads
  if (i < 8 * 64) xt16[(long)NROWS_T * 64 + i] = (_Float16)0.0f;
  if (i < 8 * 8)  combt[(long)NROWS_T * 8 + i] = 0.0f;
}

// ------------------------------------------------------------------ gate -----
__global__ void gate_kernel(const float* __restrict__ xrow, const float* __restrict__ wg,
                            float* __restrict__ comb, int rows) {
  int n = blockIdx.x * blockDim.x + threadIdx.x;
  if (n >= rows) return;
  const float* xr = xrow + (long)n * 64;
  float lg[EE];
  for (int e = 0; e < EE; e++) {
    float s = 0.f;
    for (int k = 0; k < 64; k++) s += xr[k] * wg[k * EE + e];
    lg[e] = s;
  }
  float mx = lg[0];
  for (int e = 1; e < EE; e++) mx = fmaxf(mx, lg[e]);
  float sum = 0.f;
  float sc[EE];
  for (int e = 0; e < EE; e++) { sc[e] = expf(lg[e] - mx); sum += sc[e]; }
  for (int e = 0; e < EE; e++) sc[e] /= sum;
  int i0 = 0;
  for (int e = 1; e < EE; e++) if (sc[e] > sc[i0]) i0 = e;
  int i1 = (i0 == 0) ? 1 : 0;
  for (int e = 0; e < EE; e++) if (e != i0 && sc[e] > sc[i1]) i1 = e;
  for (int e = 0; e < EE; e++)
    comb[(long)n * EE + e] = (e == i0) ? sc[i0] : ((e == i1) ? sc[i1] : 0.0f);
}

// ------------------------------------------------------- fused MoE + proj ----
// block = 128 threads (4 waves), one 16-row tile.  mode 0: motion (proj -> q,k_m,v_m),
// mode 1: text (proj -> k_t,v_t).
__global__ void moe_kernel(const _Float16* __restrict__ xf16, const float* __restrict__ comb,
                           const _Float16* __restrict__ pw1, const _Float16* __restrict__ pw2,
                           const _Float16* __restrict__ pproj,
                           const float* __restrict__ b1, const float* __restrict__ b2,
                           const float* __restrict__ projb, const float* __restrict__ kscale,
                           _Float16* __restrict__ qf, _Float16* __restrict__ kpad,
                           _Float16* __restrict__ vpadT, int mode) {
  const int tid = threadIdx.x;
  const int w = tid >> 5, lane = tid & 31, lh = lane >> 4, ln = lane & 15;
  const int row0 = blockIdx.x * 16;
  __shared__ __align__(16) _Float16 xls[16 * 64];
  __shared__ __align__(16) _Float16 hls[16 * 256];
  __shared__ __align__(16) _Float16 yls[16 * 64];
  __shared__ float cls[16 * 8];

  // async-stage the 16x64 X tile (2 KB) once per block: 128 lanes x 16 B
  const _Float16* abase = xf16 + (long)row0 * 64;
  async_copy_b128(abase + tid * 8, &xls[tid * 8]);
  if (tid < 128) cls[tid] = comb[(long)row0 * 8 + tid];
  async_wait0();
  __syncthreads();

  const v8f vz = {0.f,0.f,0.f,0.f,0.f,0.f,0.f,0.f};
  v8f yacc = vz;

  for (int e = 0; e < EE; e++) {
    // GEMM1: X(16x64) @ W1[e](64x256), wave w owns n-tiles 4w..4w+3
    v8f c0 = vz, c1 = vz, c2 = vz, c3 = vz;
    for (int s = 0; s < 2; s++) {
      __builtin_prefetch(pw1 + ((long)(e * 32 + (s + 1) * 16 + w * 4)) * 512, 0, 0);
      v16h a = load_a16(xls, 64, s * 32);
      c0 = wmma16(a, load_bpacked(pw1 + ((long)(e * 32 + s * 16 + w * 4 + 0)) * 512), c0);
      c1 = wmma16(a, load_bpacked(pw1 + ((long)(e * 32 + s * 16 + w * 4 + 1)) * 512), c1);
      c2 = wmma16(a, load_bpacked(pw1 + ((long)(e * 32 + s * 16 + w * 4 + 2)) * 512), c2);
      c3 = wmma16(a, load_bpacked(pw1 + ((long)(e * 32 + s * 16 + w * 4 + 3)) * 512), c3);
    }
    v8f cc[4] = {c0, c1, c2, c3};
    for (int t = 0; t < 4; t++) {
      int col = (w * 4 + t) * 16 + ln;
      for (int r = 0; r < 8; r++) {
        int m = r + 8 * lh;
        hls[m * 256 + col] = (_Float16)gelu_f(cc[t][r] + b1[e * 256 + col]);
      }
    }
    __syncthreads();
    // GEMM2: gelu(H)(16x256) @ W2[e](256x64), wave w owns out cols 16w..16w+15
    v8f t2 = vz;
    for (int s = 0; s < 8; s++) {
      __builtin_prefetch(pw2 + ((long)(e * 32 + (s + 1) * 4 + w)) * 512, 0, 0);
      v16h a = load_a16(hls, 256, s * 32);
      t2 = wmma16(a, load_bpacked(pw2 + ((long)(e * 32 + s * 4 + w)) * 512), t2);
    }
    int col = w * 16 + ln;
    for (int r = 0; r < 8; r++) {
      int m = r + 8 * lh;
      yacc[r] += (t2[r] + b2[e * 64 + col]) * cls[m * 8 + e];
    }
    __syncthreads();   // protect hls before next expert rewrites it
  }

  // y -> gelu -> LDS (16x64 f16)
  {
    int col = w * 16 + ln;
    for (int r = 0; r < 8; r++) {
      int m = r + 8 * lh;
      yls[m * 64 + col] = (_Float16)gelu_f(yacc[r]);
    }
  }
  __syncthreads();

  // GEMM3: gelu(y)(16x64) @ proj(64 x {192|128} used cols)
  const int NT3 = (mode == 0) ? 12 : 8;     // used n-tiles
  const int NTfull = (mode == 0) ? 20 : 8;  // packed row of blobs
  const float scl = kscale[0];
  for (int nt = w; nt < NT3; nt += 4) {
    v8f c3 = vz;
    for (int s = 0; s < 2; s++) {
      v16h a = load_a16(yls, 64, s * 32);
      c3 = wmma16(a, load_bpacked(pproj + ((long)(s * NTfull + nt)) * 512), c3);
    }
    int col = nt * 16 + ln;
    int chunk = col >> 6, d = col & 63;
    for (int r = 0; r < 8; r++) {
      int n = row0 + r + 8 * lh;
      float v = c3[r] + projb[col];
      if (mode == 0) {
        int b = n / (TT * HH); int rem = n % (TT * HH);
        int t = rem / HH, h = rem % HH;
        long bh = (long)(b * HH + h);
        if (chunk == 0)      qf[(bh * TT + t) * 64 + d]      = (_Float16)(v * 0.125f); // fold D^-1/2
        else if (chunk == 1) kpad[(bh * NKPAD + t) * 64 + d] = (_Float16)(v * scl);
        else                 vpadT[(bh * 64 + d) * NKPAD + t] = (_Float16)v;
      } else {
        if (n < NROWS_T) {
          int b = n / (MMTOK * HH); int rem = n % (MMTOK * HH);
          int mt = rem / HH, h = rem % HH;
          long bh = (long)(b * HH + h);
          int nn = 608 + mt;
          if (chunk == 0) kpad[(bh * NKPAD + nn) * 64 + d]   = (_Float16)(v * scl);
          else            vpadT[(bh * 64 + d) * NKPAD + nn]  = (_Float16)v;
        }
      }
    }
  }
}

// ------------------------------------------- dataset/rotation/pad K,V fill ---
__global__ void fill_kv_kernel(const float* __restrict__ kds, const float* __restrict__ vds,
                               const float* __restrict__ krot, const float* __restrict__ vrot,
                               const float* __restrict__ kds_s, const float* __restrict__ krot_s,
                               _Float16* __restrict__ kpad, _Float16* __restrict__ vpadT) {
  long i = (long)blockIdx.x * blockDim.x + threadIdx.x;
  long total = (long)BB * HH * (NKPAD - 512) * 64;
  if (i >= total) return;
  int d = (int)(i & 63);
  long r = i >> 6;
  int nn = 512 + (int)(r % (NKPAD - 512));
  long bh = r / (NKPAD - 512);
  int h = (int)(bh % HH);
  float kv = 0.f, vv = 0.f;
  if (nn < 560)       { int s = nn - 512; kv = kds[(s * HH + h) * 64 + d] * kds_s[0];
                        vv = vds[(s * HH + h) * 64 + d]; }
  else if (nn < 608)  { int s = nn - 560; kv = krot[(s * HH + h) * 64 + d] * krot_s[0];
                        vv = vrot[(s * HH + h) * 64 + d]; }
  else if (nn < 685)  return;   // text region written by moe_kernel
  kpad[(bh * NKPAD + nn) * 64 + d] = (_Float16)kv;
  vpadT[(bh * 64 + d) * NKPAD + nn] = (_Float16)vv;
}

__global__ void maskbias_kernel(const int* __restrict__ src_mask, const float* __restrict__ text_cond,
                                float* __restrict__ mb) {
  int i = blockIdx.x * blockDim.x + threadIdx.x;
  if (i >= BB * NKPAD) return;
  int b = i / NKPAD, n = i % NKPAD;
  float v;
  if (n < 512)      v = (src_mask[b * TT + n] > 0) ? 0.0f : -1e9f;
  else if (n < 608) v = 0.0f;
  else if (n < 685) v = (text_cond[b] > 0.0f) ? 0.0f : -1e9f;
  else              v = -1e9f;
  mb[i] = v;
}

// -------------------------------------------------- flash attention (wave) ---
// grid (T/16, B*H), block 32.  Online-softmax over 22 key tiles of 32.
__global__ void attn_kernel(const _Float16* __restrict__ qf, const _Float16* __restrict__ kpad,
                            const _Float16* __restrict__ vpadT, const float* __restrict__ mb,
                            const float* __restrict__ sigma_p, float* __restrict__ aout) {
  const int lane = threadIdx.x & 31, lh = lane >> 4, ln = lane & 15;
  const int qt = blockIdx.x;
  const int bh = blockIdx.y;
  const int b = bh / HH, h = bh % HH;
  __shared__ __align__(16) _Float16 pls[16 * 32];

  const _Float16* qbase = qf + ((long)bh * TT + qt * 16) * 64;
  const _Float16* kbase = kpad + (long)bh * NKPAD * 64;
  const _Float16* vbase = vpadT + (long)bh * 64 * NKPAD;
  const float sg = sigma_p[0];
  const float inv2s = 1.0f / (2.0f * sg * sg);

  v16h aq0 = load_a16(qbase, 64, 0);
  v16h aq1 = load_a16(qbase, 64, 32);

  const v8f vz = {0.f,0.f,0.f,0.f,0.f,0.f,0.f,0.f};
  v8f o[4] = {vz, vz, vz, vz};
  float mrow[8], lrow[8];
  for (int r = 0; r < 8; r++) { mrow[r] = -3.0e38f; lrow[r] = 0.0f; }

  for (int kt = 0; kt < NKTILES; kt++) {
    const _Float16* kb = kbase + (long)kt * 32 * 64;
    __builtin_prefetch(kbase + (long)(kt + 1) * 32 * 64, 0, 0);
    v8f s0 = vz, s1 = vz;
    s0 = wmma16(aq0, load_b16(kb, 64, 0, 0),   s0);
    s0 = wmma16(aq1, load_b16(kb, 64, 0, 32),  s0);
    s1 = wmma16(aq0, load_b16(kb, 64, 16, 0),  s1);
    s1 = wmma16(aq1, load_b16(kb, 64, 16, 32), s1);

    int n0 = kt * 32 + ln, n1 = n0 + 16;
    float bias0 = mb[b * NKPAD + n0];
    float bias1 = mb[b * NKPAD + n1];

    for (int r = 0; r < 8; r++) {
      int trow = qt * 16 + r + 8 * lh;
      float dt0 = (float)(trow - n0), dt1 = (float)(trow - n1);
      float v0 = s0[r] + bias0 - ((n0 < 512) ? dt0 * dt0 * inv2s : 0.0f);
      float v1 = s1[r] + bias1 - ((n1 < 512) ? dt1 * dt1 * inv2s : 0.0f);
      float mx = redmax16(fmaxf(v0, v1));
      float nm = fmaxf(mrow[r], mx);
      float alpha = expf(mrow[r] - nm);
      float e0 = expf(v0 - nm), e1 = expf(v1 - nm);
      lrow[r] = lrow[r] * alpha + redsum16(e0 + e1);
      mrow[r] = nm;
      for (int t = 0; t < 4; t++) o[t][r] *= alpha;
      pls[(r + 8 * lh) * 32 + ln]      = (_Float16)e0;
      pls[(r + 8 * lh) * 32 + 16 + ln] = (_Float16)e1;
    }
    __syncthreads();
    v16h ap = load_a16(pls, 32, 0);
    const _Float16* vb = vbase + kt * 32;
    for (int t = 0; t < 4; t++)
      o[t] = wmma16(ap, load_b16(vb, NKPAD, t * 16, 0), o[t]);
    __syncthreads();
  }

  for (int t = 0; t < 4; t++)
    for (int r = 0; r < 8; r++) {
      int row = qt * 16 + r + 8 * lh;
      aout[((long)b * TT + row) * LAT + h * 64 + t * 16 + ln] = o[t][r] / lrow[r];
    }
}

// ------------------------------------------------------------- output path ---
__global__ void silu_f16_kernel(const float* __restrict__ in, _Float16* __restrict__ out, long n) {
  long stride = (long)gridDim.x * blockDim.x;
  for (long i = (long)blockIdx.x * blockDim.x + threadIdx.x; i < n; i += stride)
    out[i] = (_Float16)silu_f(in[i]);
}

// eo = silu(emb)(2048x512) @ st_emb_w(512x1280) + b.  grid (128, 20), block 128.
__global__ void gemm_eo_kernel(const _Float16* __restrict__ A, const _Float16* __restrict__ pw,
                               const float* __restrict__ bias, float* __restrict__ out) {
  const int w = threadIdx.x >> 5, lane = threadIdx.x & 31, lh = lane >> 4, ln = lane & 15;
  const int row0 = blockIdx.x * 16;
  const int nt = blockIdx.y * 4 + w;
  __shared__ __align__(16) _Float16 als[16 * TED];   // 16 KB A tile
  const _Float16* ab = A + (long)row0 * TED;
  // async-stage A tile: 1024 chunks of 16 B, 8 per thread
  for (int it = 0; it < 8; it++) {
    int idx = it * 128 + threadIdx.x;
    async_copy_b128(ab + idx * 8, &als[idx * 8]);
  }
  async_wait0();
  __syncthreads();

  const v8f vz = {0.f,0.f,0.f,0.f,0.f,0.f,0.f,0.f};
  v8f c = vz;
  for (int s = 0; s < 16; s++) {
    __builtin_prefetch(pw + ((long)((s + 1) * 80 + nt)) * 512, 0, 0);
    v16h a = load_a16(als, TED, s * 32);
    c = wmma16(a, load_bpacked(pw + ((long)(s * 80 + nt)) * 512), c);
  }
  int col = nt * 16 + ln;
  for (int r = 0; r < 8; r++) {
    int row = row0 + r + 8 * lh;
    out[(long)row * 1280 + col] = c[r] + bias[col];
  }
}

// d_out = x + silu(hh)(2048x640) @ st_out_w(640x640) + b.  grid (128, 10), block 128.
__global__ void gemm_out_kernel(const _Float16* __restrict__ A, const _Float16* __restrict__ pw,
                                const float* __restrict__ bias, const float* __restrict__ x,
                                float* __restrict__ out) {
  const int w = threadIdx.x >> 5, lane = threadIdx.x & 31, lh = lane >> 4, ln = lane & 15;
  const int row0 = blockIdx.x * 16;
  const int nt = blockIdx.y * 4 + w;
  __shared__ __align__(16) _Float16 als[16 * LAT];   // 20 KB A tile
  const _Float16* ab = A + (long)row0 * LAT;
  // async-stage A tile: 1280 chunks of 16 B, 10 per thread
  for (int it = 0; it < 10; it++) {
    int idx = it * 128 + threadIdx.x;
    async_copy_b128(ab + idx * 8, &als[idx * 8]);
  }
  async_wait0();
  __syncthreads();

  const v8f vz = {0.f,0.f,0.f,0.f,0.f,0.f,0.f,0.f};
  v8f c = vz;
  for (int s = 0; s < 20; s++) {
    __builtin_prefetch(pw + ((long)((s + 1) * 40 + nt)) * 512, 0, 0);
    v16h a = load_a16(als, LAT, s * 32);
    c = wmma16(a, load_bpacked(pw + ((long)(s * 40 + nt)) * 512), c);
  }
  int col = nt * 16 + ln;
  for (int r = 0; r < 8; r++) {
    long row = row0 + r + 8 * lh;
    out[row * LAT + col] = x[row * LAT + col] + c[r] + bias[col];
  }
}

// =============================================================== launcher ====
extern "C" void kernel_launch(void* const* d_in, const int* in_sizes, int n_in,
                              void* d_out, int out_size, void* d_ws, size_t ws_size,
                              hipStream_t stream) {
  // inputs (setup_inputs order)
  const float* x           = (const float*)d_in[0];
  const float* emb         = (const float*)d_in[1];
  const int*   src_mask    = (const int*)  d_in[2];
  const float* text_cond   = (const float*)d_in[5];
  const float* text_word   = (const float*)d_in[6];
  const float* norm_g      = (const float*)d_in[7];
  const float* norm_b      = (const float*)d_in[8];
  const float* moe_emb     = (const float*)d_in[9];
  const float* m_wg        = (const float*)d_in[10];
  const float* m_w1        = (const float*)d_in[11];
  const float* m_b1        = (const float*)d_in[12];
  const float* m_w2        = (const float*)d_in[13];
  const float* m_b2        = (const float*)d_in[14];
  const float* m_proj_w    = (const float*)d_in[15];
  const float* m_proj_b    = (const float*)d_in[16];
  const float* kms         = (const float*)d_in[17];
  const float* sigma       = (const float*)d_in[18];
  const float* key_dataset = (const float*)d_in[19];
  const float* kds_s       = (const float*)d_in[20];
  const float* val_dataset = (const float*)d_in[21];
  const float* key_rot     = (const float*)d_in[22];
  const float* krot_s      = (const float*)d_in[23];
  const float* val_rot     = (const float*)d_in[24];
  const float* normt_g     = (const float*)d_in[25];
  const float* normt_b     = (const float*)d_in[26];
  const float* c_wg        = (const float*)d_in[27];
  const float* c_w1        = (const float*)d_in[28];
  const float* c_b1        = (const float*)d_in[29];
  const float* c_w2        = (const float*)d_in[30];
  const float* c_b2        = (const float*)d_in[31];
  const float* c_proj_w    = (const float*)d_in[32];
  const float* c_proj_b    = (const float*)d_in[33];
  const float* kts         = (const float*)d_in[34];
  const float* st_emb_w    = (const float*)d_in[35];
  const float* st_emb_b    = (const float*)d_in[36];
  const float* st_norm_g   = (const float*)d_in[37];
  const float* st_norm_b   = (const float*)d_in[38];
  const float* st_out_w    = (const float*)d_in[39];
  const float* st_out_b    = (const float*)d_in[40];
  float* out = (float*)d_out;

  // workspace carve-out (256B aligned)
  char* p = (char*)d_ws;
  auto alloc = [&](size_t bytes) -> void* {
    void* r = (void*)p;
    p += (bytes + 255) & ~(size_t)255;
    return r;
  };
  _Float16* pw_mw1   = (_Float16*)alloc(131072 * 2);
  _Float16* pw_mw2   = (_Float16*)alloc(131072 * 2);
  _Float16* pw_mproj = (_Float16*)alloc(20480 * 2);
  _Float16* pw_cw1   = (_Float16*)alloc(131072 * 2);
  _Float16* pw_cw2   = (_Float16*)alloc(131072 * 2);
  _Float16* pw_cproj = (_Float16*)alloc(8192 * 2);
  _Float16* pw_stemb = (_Float16*)alloc(655360 * 2);
  _Float16* pw_stout = (_Float16*)alloc(409600 * 2);
  float*    xh32     = (float*)   alloc((size_t)NROWS_M * 64 * 4);
  _Float16* xh16     = (_Float16*)alloc((size_t)NROWS_M * 64 * 2);
  float*    combm    = (float*)   alloc((size_t)NROWS_M * 8 * 4);
  float*    xt32     = (float*)   alloc((size_t)NROWS_T_PAD * 64 * 4);
  _Float16* xt16     = (_Float16*)alloc((size_t)NROWS_T_PAD * 64 * 2);
  float*    combt    = (float*)   alloc((size_t)NROWS_T_PAD * 8 * 4);
  _Float16* qf       = (_Float16*)alloc((size_t)BB * HH * TT * 64 * 2);
  _Float16* kpad     = (_Float16*)alloc((size_t)BB * HH * NKPAD * 64 * 2);
  _Float16* vpadT    = (_Float16*)alloc((size_t)BB * HH * 64 * NKPAD * 2);
  float*    mbuf     = (float*)   alloc((size_t)BB * NKPAD * 4);
  float*    aout     = (float*)   alloc((size_t)BB * TT * LAT * 4);
  _Float16* embs16   = (_Float16*)alloc((size_t)BB * TT * TED * 2);
  float*    eo       = (float*)   alloc((size_t)BB * TT * 1280 * 4);
  _Float16* hh16     = (_Float16*)alloc((size_t)BB * TT * LAT * 2);

  // 1) pre-swizzle all weights into WMMA B-operand layout
  pack_b_kernel<<<512, 256, 0, stream>>>(m_w1, pw_mw1, 64, 256, 8, 64L * 256);
  pack_b_kernel<<<512, 256, 0, stream>>>(m_w2, pw_mw2, 256, 64, 8, 256L * 64);
  pack_b_kernel<<<128, 256, 0, stream>>>(m_proj_w, pw_mproj, 64, 320, 1, 0);
  pack_b_kernel<<<512, 256, 0, stream>>>(c_w1, pw_cw1, 64, 256, 8, 64L * 256);
  pack_b_kernel<<<512, 256, 0, stream>>>(c_w2, pw_cw2, 256, 64, 8, 256L * 64);
  pack_b_kernel<<<64, 256, 0, stream>>>(c_proj_w, pw_cproj, 64, 128, 1, 0);
  pack_b_kernel<<<1024, 256, 0, stream>>>(st_emb_w, pw_stemb, 512, 1280, 1, 0);
  pack_b_kernel<<<1024, 256, 0, stream>>>(st_out_w, pw_stout, 640, 640, 1, 0);

  // 2) layernorms (+moe positional emb) -> f32 (gating) + f16 (WMMA A)
  ln_kernel<<<BB * TT, 256, 0, stream>>>(x, norm_g, norm_b, moe_emb, xh32, xh16, LAT, TT);
  ln_kernel<<<BB * MMTOK, 256, 0, stream>>>(text_word, normt_g, normt_b, nullptr, xt32, xt16, LAT, 1);
  zero_pad_kernel<<<1, 512, 0, stream>>>(xt16, combt);

  // 3) MoE gating (top-2 softmax combine weights)
  gate_kernel<<<(NROWS_M + 127) / 128, 128, 0, stream>>>(xh32, m_wg, combm, NROWS_M);
  gate_kernel<<<(NROWS_T + 127) / 128, 128, 0, stream>>>(xt32, c_wg, combt, NROWS_T);

  // 4) fused MoE-FFN + projection -> q / k / v (f16, attention layout)
  moe_kernel<<<NROWS_M / 16, 128, 0, stream>>>(xh16, combm, pw_mw1, pw_mw2, pw_mproj,
                                               m_b1, m_b2, m_proj_b, kms,
                                               qf, kpad, vpadT, 0);
  moe_kernel<<<NROWS_T_PAD / 16, 128, 0, stream>>>(xt16, combt, pw_cw1, pw_cw2, pw_cproj,
                                                   c_b1, c_b2, c_proj_b, kts,
                                                   qf, kpad, vpadT, 1);

  // 5) dataset/rotation/pad segments + additive mask biases
  {
    long total = (long)BB * HH * (NKPAD - 512) * 64;
    fill_kv_kernel<<<(int)((total + 255) / 256), 256, 0, stream>>>(
        key_dataset, val_dataset, key_rot, val_rot, kds_s, krot_s, kpad, vpadT);
  }
  maskbias_kernel<<<(BB * NKPAD + 255) / 256, 256, 0, stream>>>(src_mask, text_cond, mbuf);

  // 6) flash attention over 704 padded keys
  attn_kernel<<<dim3(TT / 16, BB * HH), 32, 0, stream>>>(qf, kpad, vpadT, mbuf, sigma, aout);

  // 7) adaLN output path
  silu_f16_kernel<<<1024, 256, 0, stream>>>(emb, embs16, (long)BB * TT * TED);
  gemm_eo_kernel<<<dim3(BB * TT / 16, 20), 128, 0, stream>>>(embs16, pw_stemb, st_emb_b, eo);
  ln_mod_kernel<<<BB * TT, 256, 0, stream>>>(aout, st_norm_g, st_norm_b, eo, hh16);
  gemm_out_kernel<<<dim3(BB * TT / 16, 10), 128, 0, stream>>>(hh16, pw_stout, st_out_b, x, out);

  (void)in_sizes; (void)n_in; (void)out_size; (void)ws_size;
}
